// MultiHeadAttention_1056561955267
// MI455X (gfx1250) — compile-verified, matrix operand order checked by reading, not run on hardware
//
#include <hip/hip_runtime.h>
#include <hip/hip_bf16.h>
#include <math.h>

// MHA forward for B=2, S=2048, DM=1024, H=16, D=64 on gfx1250 (wave32, WMMA).
// All GEMMs use v_wmma_f32_16x16x32_f16 with 1x4 register blocking: each wave
// loads one A fragment and sweeps 4 N-tiles (4 accumulators), so the dominant
// memory stream (537 MB fp32 softmax weights) is read exactly once.

#define S_LEN   2048
#define DM      1024
#define NHEAD   16
#define DEPTH   64
#define BATCH   2
#define MROWS   (BATCH * S_LEN)      // 4096

typedef __attribute__((ext_vector_type(16))) _Float16 v16h;
typedef __attribute__((ext_vector_type(8)))  _Float16 v8h;
typedef __attribute__((ext_vector_type(8)))  float    v8f;

// ---------------------------------------------------------------------------
// WMMA wrapper: D = A(16x32 f16) * B(32x16 f16) + C(16x16 f32)
// ---------------------------------------------------------------------------
__device__ __forceinline__ v8f wmma16(v16h a, v16h b, v8f c) {
  return __builtin_amdgcn_wmma_f32_16x16x32_f16(
      /*neg_a=*/false, a, /*neg_b=*/false, b,
      /*c_mod=*/(short)0, c, /*reuse_a=*/false, /*reuse_b=*/false);
}

// 16-bit A/B fragment layout (ISA 7.12.2): lane = half*16 + r holds row r,
// K = {half*8 + 0..7, 16 + half*8 + 0..7}  ->  two contiguous 16B loads.
__device__ __forceinline__ v16h load_frag_f16(const _Float16* __restrict__ base,
                                              int ld, int row, int k0, int half) {
  const _Float16* p = base + (size_t)row * ld + k0 + half * 8;
  v8h lo = *(const v8h*)(p);
  v8h hi = *(const v8h*)(p + 16);
  v16h r;
#pragma unroll
  for (int j = 0; j < 8; ++j) { r[j] = lo[j]; r[j + 8] = hi[j]; }
  return r;
}

// Same fragment but source is fp32 (converted to f16 in registers).
__device__ __forceinline__ v16h load_frag_f32(const float* __restrict__ base,
                                              int ld, int row, int k0, int half) {
  const float* p = base + (size_t)row * ld + k0 + half * 8;
  v8f lo = *(const v8f*)(p);
  v8f hi = *(const v8f*)(p + 16);
  v16h r;
#pragma unroll
  for (int j = 0; j < 8; ++j) {
    r[j]     = (_Float16)lo[j];
    r[j + 8] = (_Float16)hi[j];
  }
  return r;
}

// ---------------------------------------------------------------------------
// Kernel 1: convert + transpose a DMxDM fp32 weight matrix to f16 (N x K).
// ---------------------------------------------------------------------------
__global__ void __launch_bounds__(256)
wcvt_kernel(const float* __restrict__ W, _Float16* __restrict__ Wt) {
  int idx = blockIdx.x * 256 + threadIdx.x;      // over DM*DM
  int n = idx >> 10;
  int k = idx & (DM - 1);
  Wt[idx] = (_Float16)W[(size_t)k * DM + n];
}

// ---------------------------------------------------------------------------
// Kernel 2: projection GEMM.  P = X(M x K fp32) * Wt^T + bias, stored f16 in
// per-head layout: base (b*H+h)*S*D + s*s_stride + d*d_stride.
//   q,k: s_stride=D, d_stride=1       (B,H,S,D)
//   v:   s_stride=1, d_stride=S       (B,H,D,S)  -> B-operand of PV is K-major
// Block = 4 waves (one M-tile each); each wave sweeps 4 N-tiles per A load.
// ---------------------------------------------------------------------------
__global__ void __launch_bounds__(128)
proj_gemm_kernel(const float* __restrict__ X, const _Float16* __restrict__ Wt,
                 const float* __restrict__ bias, _Float16* __restrict__ outp,
                 int s_stride, int d_stride) {
  const int lane = threadIdx.x & 31;
  const int wave = threadIdx.x >> 5;
  const int half = lane >> 4;
  const int r    = lane & 15;
  const int row0 = (blockIdx.y * 4 + wave) * 16;     // M tile (per wave)
  const int col0 = blockIdx.x * 64;                  // N tile group

  v8f acc[4] = {};
#pragma unroll 2
  for (int k0 = 0; k0 < DM; k0 += 32) {
    v16h a = load_frag_f32(X, DM, row0 + r, k0, half);
#pragma unroll
    for (int t = 0; t < 4; ++t) {
      v16h b = load_frag_f16(Wt, DM, col0 + 16 * t + r, k0, half);
      acc[t] = wmma16(a, b, acc[t]);
    }
  }

#pragma unroll
  for (int t = 0; t < 4; ++t) {
    const int dm = col0 + 16 * t + r;    // output column for this lane
    const float bv = bias[dm];
    const int h = dm >> 6;
    const int d = dm & (DEPTH - 1);
#pragma unroll
    for (int i = 0; i < 8; ++i) {
      int row  = row0 + i + 8 * half;    // global row in [0, B*S)
      int bidx = row >> 11;              // / S
      int s    = row & (S_LEN - 1);
      float val = acc[t][i] + bv;
      outp[(size_t)(bidx * NHEAD + h) * (S_LEN * DEPTH)
           + (size_t)s * s_stride + (size_t)d * d_stride] = (_Float16)val;
    }
  }
}

// ---------------------------------------------------------------------------
// Kernel 3: logits = q . k^T * (1/sqrt(D)), fp32 into the weights region.
// q,k: (B*H, S, D) f16.  B-operand = k^T so both fragments are K-major.
// Each wave: one q row tile, 4 k col tiles.
// ---------------------------------------------------------------------------
__global__ void __launch_bounds__(128)
logits_kernel(const _Float16* __restrict__ qh, const _Float16* __restrict__ kh,
              float* __restrict__ wts) {
  const int lane = threadIdx.x & 31;
  const int wave = threadIdx.x >> 5;
  const int half = lane >> 4;
  const int r    = lane & 15;
  const int bh    = blockIdx.z;
  const int qrow0 = (blockIdx.y * 4 + wave) * 16;
  const int krow0 = blockIdx.x * 64;

  const _Float16* qp = qh + (size_t)bh * S_LEN * DEPTH;
  const _Float16* kp = kh + (size_t)bh * S_LEN * DEPTH;

  v8f acc[4] = {};
#pragma unroll
  for (int k0 = 0; k0 < DEPTH; k0 += 32) {
    v16h a = load_frag_f16(qp, DEPTH, qrow0 + r, k0, half);
#pragma unroll
    for (int t = 0; t < 4; ++t) {
      v16h b = load_frag_f16(kp, DEPTH, krow0 + 16 * t + r, k0, half);
      acc[t] = wmma16(a, b, acc[t]);
    }
  }

  float* wrow = wts + (size_t)bh * S_LEN * S_LEN;
#pragma unroll
  for (int t = 0; t < 4; ++t) {
#pragma unroll
    for (int i = 0; i < 8; ++i) {
      int qi = qrow0 + i + 8 * half;
      wrow[(size_t)qi * S_LEN + krow0 + 16 * t + r] =
          acc[t][i] * 0.125f;            // 1/sqrt(64)
    }
  }
}

// ---------------------------------------------------------------------------
// Kernel 4: in-place row softmax over 2048 columns. One block per row.
// ---------------------------------------------------------------------------
__global__ void __launch_bounds__(256)
softmax_kernel(float* __restrict__ wts) {
  __shared__ float red[256];
  float* p = wts + (size_t)blockIdx.x * S_LEN;
  const int t = threadIdx.x;

  float vals[8];
  float m = -3.0e38f;
#pragma unroll
  for (int i = 0; i < 8; ++i) {
    vals[i] = p[t + i * 256];
    m = fmaxf(m, vals[i]);
  }
  red[t] = m;
  __syncthreads();
  for (int s = 128; s > 0; s >>= 1) {
    if (t < s) red[t] = fmaxf(red[t], red[t + s]);
    __syncthreads();
  }
  m = red[0];
  __syncthreads();

  float sum = 0.f;
#pragma unroll
  for (int i = 0; i < 8; ++i) {
    vals[i] = __expf(vals[i] - m);
    sum += vals[i];
  }
  red[t] = sum;
  __syncthreads();
  for (int s = 128; s > 0; s >>= 1) {
    if (t < s) red[t] += red[t + s];
    __syncthreads();
  }
  const float inv = 1.0f / red[0];
#pragma unroll
  for (int i = 0; i < 8; ++i) p[t + i * 256] = vals[i] * inv;
}

// ---------------------------------------------------------------------------
// Kernel 5: attn = weights(fp32, cvt inline) x v. v stored (B*H, D, S) f16 so
// the B fragment is K-major too. Each wave owns one query row tile and sweeps
// all 4 d-tiles (DEPTH=64) per A load -> the 537 MB weights tensor is read
// exactly once. Output merged-head (B, S, DM) f16.
// ---------------------------------------------------------------------------
__global__ void __launch_bounds__(128)
attnv_kernel(const float* __restrict__ wts, const _Float16* __restrict__ vt,
             _Float16* __restrict__ attn16) {
  const int lane = threadIdx.x & 31;
  const int wave = threadIdx.x >> 5;
  const int half = lane >> 4;
  const int r    = lane & 15;
  const int bh   = blockIdx.z;
  const int row0 = (blockIdx.y * 4 + wave) * 16;   // query rows (per wave)

  const float*    wp = wts + (size_t)bh * S_LEN * S_LEN;
  const _Float16* vp = vt  + (size_t)bh * S_LEN * DEPTH;  // (D x S)

  const float* arow = wp + (size_t)(row0 + r) * S_LEN + half * 8;

  v8f acc[4] = {};
#pragma unroll 2
  for (int k0 = 0; k0 < S_LEN; k0 += 32) {
    // prefetch the A stream ~8 iterations ahead (global_prefetch_b8)
    __builtin_prefetch(arow + k0 + 256, 0, 1);
    v16h a = load_frag_f32(wp, S_LEN, row0 + r, k0, half);
#pragma unroll
    for (int t = 0; t < 4; ++t) {
      v16h b = load_frag_f16(vp, S_LEN, 16 * t + r, k0, half);
      acc[t] = wmma16(a, b, acc[t]);
    }
  }

  const int b = bh >> 4;
  const int h = bh & (NHEAD - 1);
#pragma unroll
  for (int t = 0; t < 4; ++t) {
#pragma unroll
    for (int i = 0; i < 8; ++i) {
      int s = row0 + i + 8 * half;
      attn16[((size_t)(b * S_LEN + s)) * DM + h * DEPTH + 16 * t + r] =
          (_Float16)acc[t][i];
    }
  }
}

// ---------------------------------------------------------------------------
// Kernel 6: out = attn(f16) x Wo^T(f16) + bo, fp32 output.
// ---------------------------------------------------------------------------
__global__ void __launch_bounds__(128)
out_gemm_kernel(const _Float16* __restrict__ A, const _Float16* __restrict__ Wot,
                const float* __restrict__ bo, float* __restrict__ out) {
  const int lane = threadIdx.x & 31;
  const int wave = threadIdx.x >> 5;
  const int half = lane >> 4;
  const int r    = lane & 15;
  const int row0 = (blockIdx.y * 4 + wave) * 16;
  const int col0 = blockIdx.x * 64;

  v8f acc[4] = {};
#pragma unroll 2
  for (int k0 = 0; k0 < DM; k0 += 32) {
    v16h a = load_frag_f16(A, DM, row0 + r, k0, half);
#pragma unroll
    for (int t = 0; t < 4; ++t) {
      v16h b = load_frag_f16(Wot, DM, col0 + 16 * t + r, k0, half);
      acc[t] = wmma16(a, b, acc[t]);
    }
  }

#pragma unroll
  for (int t = 0; t < 4; ++t) {
    const float bv = bo[col0 + 16 * t + r];
#pragma unroll
    for (int i = 0; i < 8; ++i) {
      int row = row0 + i + 8 * half;
      out[(size_t)row * DM + col0 + 16 * t + r] = acc[t][i] + bv;
    }
  }
}

// ---------------------------------------------------------------------------
extern "C" void kernel_launch(void* const* d_in, const int* in_sizes, int n_in,
                              void* d_out, int out_size, void* d_ws, size_t ws_size,
                              hipStream_t stream) {
  const float* Q  = (const float*)d_in[0];
  const float* K  = (const float*)d_in[1];
  const float* V  = (const float*)d_in[2];
  const float* Wq = (const float*)d_in[3];
  const float* bq = (const float*)d_in[4];
  const float* Wk = (const float*)d_in[5];
  const float* bk = (const float*)d_in[6];
  const float* Wv = (const float*)d_in[7];
  const float* bv = (const float*)d_in[8];
  const float* Wo = (const float*)d_in[9];
  const float* bo = (const float*)d_in[10];

  float* out = (float*)d_out;                       // (B,S,DM) fp32
  float* wts = out + (size_t)MROWS * DM;            // (B,H,S,S) fp32

  const size_t MAT = (size_t)DM * DM;               // 1M elements
  _Float16* ws     = (_Float16*)d_ws;
  _Float16* Wqt    = ws + 0 * MAT;
  _Float16* Wkt    = ws + 1 * MAT;
  _Float16* Wvt    = ws + 2 * MAT;
  _Float16* Wot    = ws + 3 * MAT;
  _Float16* qh     = ws + 4 * MAT;                  // (B,H,S,D) f16
  _Float16* kh     = ws + 8 * MAT;                  // (B,H,S,D) f16
  _Float16* vt     = ws + 12 * MAT;                 // (B,H,D,S) f16
  _Float16* attn16 = ws + 16 * MAT;                 // (B,S,DM) f16

  dim3 b256(256), b128(128);

  // 1) weight convert + transpose (f32 -> f16, N x K)
  wcvt_kernel<<<dim3(DM * DM / 256), b256, 0, stream>>>(Wq, Wqt);
  wcvt_kernel<<<dim3(DM * DM / 256), b256, 0, stream>>>(Wk, Wkt);
  wcvt_kernel<<<dim3(DM * DM / 256), b256, 0, stream>>>(Wv, Wvt);
  wcvt_kernel<<<dim3(DM * DM / 256), b256, 0, stream>>>(Wo, Wot);

  // 2) projections (WMMA): block covers 64x64 C tile
  dim3 gProj(DM / 64, MROWS / 64);                  // (16, 64)
  proj_gemm_kernel<<<gProj, b128, 0, stream>>>(Q, Wqt, bq, qh, DEPTH, 1);
  proj_gemm_kernel<<<gProj, b128, 0, stream>>>(K, Wkt, bk, kh, DEPTH, 1);
  proj_gemm_kernel<<<gProj, b128, 0, stream>>>(V, Wvt, bv, vt, 1, S_LEN);

  // 3) logits (WMMA) -> weights region of d_out
  logits_kernel<<<dim3(S_LEN / 64, S_LEN / 64, BATCH * NHEAD), b128, 0, stream>>>(
      qh, kh, wts);

  // 4) softmax in place (one block per row)
  softmax_kernel<<<dim3(BATCH * NHEAD * S_LEN), b256, 0, stream>>>(wts);

  // 5) attn = weights x v (WMMA, weights stream read exactly once)
  attnv_kernel<<<dim3(1, S_LEN / 64, BATCH * NHEAD), b128, 0, stream>>>(
      wts, vt, attn16);

  // 6) output projection (WMMA)
  out_gemm_kernel<<<gProj, b128, 0, stream>>>(attn16, Wot, bo, out);
}
